// DecoderLayer_52527450030552
// MI455X (gfx1250) — matrix operand order repaired and checked by reading, not compiled, not run on hardware
//
#include <hip/hip_runtime.h>

// ---------------------------------------------------------------------------
// CDNA5 (gfx1250, wave32) decoder layer. bf16 WMMA GEMMs with fp32 accumulate,
// flash cross-attention (no materialized 512MB score tensor), L2-resident
// bf16 weights loaded directly as WMMA fragments (no LDS staging for GEMMs).
// Round 2: flash grid reordered so batch is innermost (the 128MB cross bias is
// batch-independent -> 4x L2 reuse instead of 4x HBM re-read), global_prefetch
// for bias/A streams, vectorized bf16 loads in the small self-attn kernel.
// ---------------------------------------------------------------------------

typedef __attribute__((ext_vector_type(16))) __bf16 v16bf;
typedef __attribute__((ext_vector_type(8)))  __bf16 bf16x8;
typedef __attribute__((ext_vector_type(8)))  float  v8f;

union AFrag { v16bf v; bf16x8 h[2]; };

// ---------------------- elementwise / conversion kernels --------------------

// Convert fp32 weight (K x N, row-major) -> bf16 transposed (N x K, row-major)
__global__ __launch_bounds__(256) void convT_kernel(const float* __restrict__ W,
                                                    __bf16* __restrict__ T,
                                                    int K, int N) {
    size_t idx = (size_t)blockIdx.x * 256 + threadIdx.x;
    if (idx >= (size_t)K * N) return;
    int n = (int)(idx / (size_t)K);
    int k = (int)(idx % (size_t)K);
    T[idx] = (__bf16)W[(size_t)k * N + n];
}

__global__ __launch_bounds__(256) void conv_kernel(const float* __restrict__ X,
                                                   __bf16* __restrict__ Y, size_t n) {
    size_t i = (size_t)blockIdx.x * 256 + threadIdx.x;
    if (i < n) Y[i] = (__bf16)X[i];
}

// RMSNorm over last dim (1024), out bf16. One block (256 thr) per row.
__global__ __launch_bounds__(256) void rmsnorm_kernel(const float* __restrict__ x,
                                                      const float* __restrict__ w,
                                                      __bf16* __restrict__ out) {
    __shared__ float red[256];
    int tid = threadIdx.x;
    const float* xr = x + (size_t)blockIdx.x * 1024;
    float vals[4];
    float s = 0.f;
    #pragma unroll
    for (int j = 0; j < 4; j++) {
        vals[j] = xr[tid + j * 256];
        s += vals[j] * vals[j];
    }
    red[tid] = s;
    __syncthreads();
    for (int off = 128; off > 0; off >>= 1) {
        if (tid < off) red[tid] += red[tid + off];
        __syncthreads();
    }
    float rstd = rsqrtf(red[0] * (1.0f / 1024.0f) + 1e-6f);
    __bf16* orow = out + (size_t)blockIdx.x * 1024;
    #pragma unroll
    for (int j = 0; j < 4; j++) {
        int c = tid + j * 256;
        orow[c] = (__bf16)(vals[j] * rstd * w[c]);
    }
}

// x1 = x + broadcast(selfproj[b, :]) over all 2048 positions (ref broadcasts!)
__global__ __launch_bounds__(256) void bcast_add_kernel(const float* __restrict__ x,
                                                        const float* __restrict__ sp,
                                                        float* __restrict__ y) {
    size_t i = (size_t)blockIdx.x * 256 + threadIdx.x;   // 2^23 elements
    int c = (int)(i & 1023);
    int b = (int)(i >> 21);                              // (i/1024)/2048
    y[i] = x[i] + sp[b * 1024 + c];
}

// ----------------------------- WMMA GEMM ------------------------------------
// C(M,N) = A(M,K,bf16 row-major) * B^T(N,K,bf16 row-major).
// Block: 256 thr = 8 waves; wave w -> rows [row0, row0+16), cols [col0, col0+64).
// A frag: per lane 2x16B contiguous reads; B frag: per lane 32B contiguous read
// from the pre-transposed weight (all bf16 weights fit in the 192MB L2).
// MODE: 0 = f32 out, 1 = bf16 out, 2 = bf16 relu out, 3 = f32 residual-add out.
template <int MODE, bool TOUT>
__global__ __launch_bounds__(256) void gemm_kernel(const __bf16* __restrict__ A,
                                                   const __bf16* __restrict__ BT,
                                                   void* __restrict__ Cout,
                                                   const float* __restrict__ Res,
                                                   int M, int N, int K, int mb) {
    int lane = threadIdx.x & 31;
    int wv   = threadIdx.x >> 5;
    int lrow = lane & 15;
    int hi   = lane >> 4;
    int kA0  = hi ? 16 : 0;
    int kB0  = hi ? 16 : 0;
    int row0 = blockIdx.x * 128 + wv * 16;
    int col0 = blockIdx.y * 64;

    int ra = row0 + lrow;
    if (ra >= M) ra = M - 1;                 // clamp (small-M launches)
    const __bf16* arow = A + (size_t)ra * K;

    v8f acc[4] = {{}, {}, {}, {}};
    for (int kb = 0; kb < K; kb += 32) {
        AFrag a;
        a.h[0] = *(const bf16x8*)(arow + kb + kA0);
        a.h[1] = *(const bf16x8*)(arow + kb + kA0 + 8);
        // speculative prefetch of the next K-step of this A row (dropped at end)
        __builtin_prefetch(arow + kb + 64, 0, 0);
        #pragma unroll
        for (int t = 0; t < 4; t++) {
            const __bf16* brow = BT + (size_t)(col0 + t * 16 + lrow) * K + kb + kB0;
            v16bf bfr = *(const v16bf*)brow;
            acc[t] = __builtin_amdgcn_wmma_f32_16x16x32_bf16(
                false, a.v, false, bfr, (short)0, acc[t], false, false);
        }
    }

    int rowoff = hi ? 8 : 0;
    #pragma unroll
    for (int v = 0; v < 8; v++) {
        int r = row0 + rowoff + v;
        if (r >= M) continue;
        #pragma unroll
        for (int t = 0; t < 4; t++) {
            int c = col0 + t * 16 + lrow;
            float val = acc[t][v];
            if (MODE == 0) {
                ((float*)Cout)[(size_t)r * N + c] = val;
            } else if (MODE == 1) {
                if (TOUT) {  // store transposed per-batch: out[b, c, r%mb]
                    int bb = r / mb, kk = r % mb;
                    ((__bf16*)Cout)[((size_t)bb * N + c) * mb + kk] = (__bf16)val;
                } else {
                    ((__bf16*)Cout)[(size_t)r * N + c] = (__bf16)val;
                }
            } else if (MODE == 2) {
                ((__bf16*)Cout)[(size_t)r * N + c] = (__bf16)fmaxf(val, 0.f);
            } else {
                size_t idx = (size_t)r * N + c;
                ((float*)Cout)[idx] = Res[idx] + val;
            }
        }
    }
}

// ------------------------- self attention (1 query) -------------------------
// Tiny workload: one block per (b,h). Computes q = normed[-1] @ wq_s head,
// scores over 2048 keys + bias, softmax, PV -> sobuf (4 x 1024 bf16).
// Dot products use 16B bf16x8 loads to avoid scalarized ushort traffic.
__global__ __launch_bounds__(256) void self_attn_kernel(const __bf16* __restrict__ nrm,
                                                        const __bf16* __restrict__ WqT,
                                                        const __bf16* __restrict__ K,
                                                        const __bf16* __restrict__ V,
                                                        const float* __restrict__ bias,
                                                        __bf16* __restrict__ so) {
    __shared__ float qh[64];
    __shared__ float sc[2048];
    __shared__ float red[256];
    __shared__ float po[4][64];
    int b = blockIdx.x >> 4, h = blockIdx.x & 15;
    int tid = threadIdx.x;

    // q head: 64 dots of length 1024 (WqT is transposed: rows are output dims)
    const __bf16* nr = nrm + ((size_t)b * 2048 + 2047) * 1024;
    {
        int d = tid >> 2, part = tid & 3;
        const bf16x8* wp = (const bf16x8*)(WqT + (size_t)(h * 64 + d) * 1024 + part * 256);
        const bf16x8* xp = (const bf16x8*)(nr + part * 256);
        float s = 0.f;
        for (int c8 = 0; c8 < 32; c8++) {
            bf16x8 xv = xp[c8], wv = wp[c8];
            #pragma unroll
            for (int j = 0; j < 8; j++) s += (float)xv[j] * (float)wv[j];
        }
        red[tid] = s;
        __syncthreads();
        if (part == 0) qh[d] = red[tid] + red[tid + 1] + red[tid + 2] + red[tid + 3];
    }
    __syncthreads();

    float lmax = -3e38f;
    for (int i = 0; i < 8; i++) {
        int n = tid + i * 256;
        const bf16x8* kp = (const bf16x8*)(K + ((size_t)b * 2048 + n) * 1024 + h * 64);
        float s = 0.f;
        #pragma unroll
        for (int c8 = 0; c8 < 8; c8++) {
            bf16x8 kv = kp[c8];
            #pragma unroll
            for (int j = 0; j < 8; j++) s += qh[c8 * 8 + j] * (float)kv[j];
        }
        s += bias[h * 2048 + n];
        sc[n] = s;
        lmax = fmaxf(lmax, s);
    }
    red[tid] = lmax;
    __syncthreads();
    for (int off = 128; off > 0; off >>= 1) {
        if (tid < off) red[tid] = fmaxf(red[tid], red[tid + off]);
        __syncthreads();
    }
    float m = red[0];
    __syncthreads();
    float ls = 0.f;
    for (int i = 0; i < 8; i++) {
        int n = tid + i * 256;
        float p = __expf(sc[n] - m);
        sc[n] = p;
        ls += p;
    }
    red[tid] = ls;
    __syncthreads();
    for (int off = 128; off > 0; off >>= 1) {
        if (tid < off) red[tid] += red[tid + off];
        __syncthreads();
    }
    float inv = 1.0f / red[0];
    __syncthreads();

    int d = tid & 63, chunk = tid >> 6;
    float acc = 0.f;
    for (int n = chunk * 512; n < chunk * 512 + 512; n++)
        acc += sc[n] * (float)V[((size_t)b * 2048 + n) * 1024 + h * 64 + d];
    po[chunk][d] = acc;
    __syncthreads();
    if (tid < 64)
        so[(size_t)b * 1024 + h * 64 + tid] =
            (__bf16)((po[0][tid] + po[1][tid] + po[2][tid] + po[3][tid]) * inv);
}

// ------------------------- flash cross-attention ----------------------------
// grid = 1024 blocks. BLOCK ORDER: batch is the INNERMOST index — the 128MB
// bias tensor is batch-independent, so the 4 consecutive blocks sharing one
// (h, q-chunk) bias slice (512KB) hit L2 instead of re-streaming 4x from HBM
// (saves ~384MB of the ~900MB total HBM traffic). K/V head slices (8MB total)
// are L2-resident under any order.
// Per 32-key step: S = Q@K^T (4 WMMA) + bias, online softmax (row stats align
// with the C-layout, shfl_xor over 16-lane halves), P relayout via per-wave
// LDS + s_wait_dscnt, O += P@V (4 WMMA from transposed V). O normalized, bf16.
__global__ __launch_bounds__(256) void flash_cross_kernel(const __bf16* __restrict__ Q,
                                                          const __bf16* __restrict__ Kc,
                                                          const __bf16* __restrict__ VT,
                                                          const float* __restrict__ bias,
                                                          __bf16* __restrict__ O) {
    __shared__ __bf16 plds[8 * 16 * 32];
    int lane  = threadIdx.x & 31;
    int w     = threadIdx.x >> 5;
    int b     = blockIdx.x & 3;            // innermost: bias L2 reuse
    int h     = (blockIdx.x >> 2) & 15;
    int chunk = blockIdx.x >> 6;
    int q0    = chunk * 128 + w * 16;
    int lrow  = lane & 15;
    int hi    = lane >> 4;
    int kA0   = hi ? 16 : 0;
    int kB0   = hi ? 16 : 0;
    int rowoff = hi ? 8 : 0;

    // hoist Q fragments (dk=64 -> two 16x32 A fragments)
    const __bf16* qbase = Q + ((size_t)b * 2048 + q0 + lrow) * 1024 + h * 64;
    AFrag qa[2];
    qa[0].h[0] = *(const bf16x8*)(qbase + 0 + kA0);
    qa[0].h[1] = *(const bf16x8*)(qbase + 8 + kA0);
    qa[1].h[0] = *(const bf16x8*)(qbase + 32 + kA0);
    qa[1].h[1] = *(const bf16x8*)(qbase + 40 + kA0);

    v8f o0 = {}, o1 = {}, o2 = {}, o3 = {};
    float vm[8], vl[8];
    #pragma unroll
    for (int v = 0; v < 8; v++) { vm[v] = -3e38f; vl[v] = 0.f; }

    __bf16* pw = plds + w * 16 * 32;

    for (int kk = 0; kk < 1024; kk += 32) {
        v8f s0 = {}, s1 = {};
        #pragma unroll
        for (int ks = 0; ks < 2; ks++) {
            const __bf16* kb = Kc + ((size_t)b * 1024 + kk) * 1024 + h * 64 + ks * 32 + kB0;
            v16bf b0 = *(const v16bf*)(kb + (size_t)lrow * 1024);
            v16bf b1 = *(const v16bf*)(kb + (size_t)(lrow + 16) * 1024);
            s0 = __builtin_amdgcn_wmma_f32_16x16x32_bf16(false, qa[ks].v, false, b0,
                                                         (short)0, s0, false, false);
            s1 = __builtin_amdgcn_wmma_f32_16x16x32_bf16(false, qa[ks].v, false, b1,
                                                         (short)0, s1, false, false);
        }
        const float* bb = bias + ((size_t)h * 2048 + q0 + rowoff) * 1024 + kk + lrow;
        // speculative prefetch of next key-step bias (dropped past tensor end)
        __builtin_prefetch(bb + 32, 0, 0);
        #pragma unroll
        for (int v = 0; v < 8; v++) {
            float a0 = s0[v] + bb[(size_t)v * 1024];
            float a1 = s1[v] + bb[(size_t)v * 1024 + 16];
            float rm = fmaxf(a0, a1);
            rm = fmaxf(rm, __shfl_xor(rm, 1, 32));
            rm = fmaxf(rm, __shfl_xor(rm, 2, 32));
            rm = fmaxf(rm, __shfl_xor(rm, 4, 32));
            rm = fmaxf(rm, __shfl_xor(rm, 8, 32));
            float mn = fmaxf(vm[v], rm);
            float alpha = __expf(vm[v] - mn);
            float p0 = __expf(a0 - mn);
            float p1 = __expf(a1 - mn);
            float rs = p0 + p1;
            rs += __shfl_xor(rs, 1, 32);
            rs += __shfl_xor(rs, 2, 32);
            rs += __shfl_xor(rs, 4, 32);
            rs += __shfl_xor(rs, 8, 32);
            vl[v] = vl[v] * alpha + rs;
            vm[v] = mn;
            o0[v] *= alpha; o1[v] *= alpha; o2[v] *= alpha; o3[v] *= alpha;
            pw[(rowoff + v) * 32 + lrow]      = (__bf16)p0;
            pw[(rowoff + v) * 32 + 16 + lrow] = (__bf16)p1;
        }
        asm volatile("s_wait_dscnt 0x0" ::: "memory");   // in-wave LDS relayout
        AFrag pa;
        pa.h[0] = *(const bf16x8*)(pw + lrow * 32 + kA0);
        pa.h[1] = *(const bf16x8*)(pw + lrow * 32 + kA0 + 8);
        const __bf16* vb = VT + ((size_t)b * 1024 + h * 64) * 1024 + kk + kB0;
        v16bf v0 = *(const v16bf*)(vb + (size_t)(lrow)      * 1024);
        v16bf v1 = *(const v16bf*)(vb + (size_t)(lrow + 16) * 1024);
        v16bf v2 = *(const v16bf*)(vb + (size_t)(lrow + 32) * 1024);
        v16bf v3 = *(const v16bf*)(vb + (size_t)(lrow + 48) * 1024);
        o0 = __builtin_amdgcn_wmma_f32_16x16x32_bf16(false, pa.v, false, v0, (short)0, o0, false, false);
        o1 = __builtin_amdgcn_wmma_f32_16x16x32_bf16(false, pa.v, false, v1, (short)0, o1, false, false);
        o2 = __builtin_amdgcn_wmma_f32_16x16x32_bf16(false, pa.v, false, v2, (short)0, o2, false, false);
        o3 = __builtin_amdgcn_wmma_f32_16x16x32_bf16(false, pa.v, false, v3, (short)0, o3, false, false);
    }

    __bf16* ob = O + ((size_t)b * 2048 + q0 + rowoff) * 1024 + h * 64 + lrow;
    #pragma unroll
    for (int v = 0; v < 8; v++) {
        float inv = 1.0f / vl[v];
        ob[(size_t)v * 1024 + 0]  = (__bf16)(o0[v] * inv);
        ob[(size_t)v * 1024 + 16] = (__bf16)(o1[v] * inv);
        ob[(size_t)v * 1024 + 32] = (__bf16)(o2[v] * inv);
        ob[(size_t)v * 1024 + 48] = (__bf16)(o3[v] * inv);
    }
}

// ------------------------------ launch --------------------------------------

extern "C" void kernel_launch(void* const* d_in, const int* in_sizes, int n_in,
                              void* d_out, int out_size, void* d_ws, size_t ws_size,
                              hipStream_t stream) {
    (void)in_sizes; (void)n_in; (void)out_size; (void)ws_size;
    const float* x      = (const float*)d_in[0];
    const float* enc    = (const float*)d_in[1];
    const float* bias_s = (const float*)d_in[2];
    const float* bias_c = (const float*)d_in[3];
    const float* Wsrc[10] = {
        (const float*)d_in[4],  (const float*)d_in[5],  (const float*)d_in[6],
        (const float*)d_in[7],  (const float*)d_in[8],  (const float*)d_in[9],
        (const float*)d_in[10], (const float*)d_in[11], (const float*)d_in[12],
        (const float*)d_in[13]
    };
    const float* n1 = (const float*)d_in[14];
    const float* n2 = (const float*)d_in[15];
    const float* n3 = (const float*)d_in[16];
    float* out = (float*)d_out;

    char* ws = (char*)d_ws;
    const size_t MB = 1024ull * 1024ull;
    // transposed bf16 weights (32MB total -> L2 resident)
    size_t woff[10] = {0, 2 * MB, 4 * MB, 6 * MB, 8 * MB, 10 * MB, 12 * MB, 14 * MB, 16 * MB, 24 * MB};
    int    wK[10]   = {1024, 1024, 1024, 1024, 1024, 1024, 1024, 1024, 1024, 4096};
    int    wN[10]   = {1024, 1024, 1024, 1024, 1024, 1024, 1024, 1024, 4096, 1024};
    __bf16* WT[10];
    for (int i = 0; i < 10; i++) WT[i] = (__bf16*)(ws + woff[i]);
    __bf16* encbf  = (__bf16*)(ws + 32 * MB);   // 8MB
    __bf16* nbuf   = (__bf16*)(ws + 40 * MB);   // 16MB
    __bf16* kbuf   = (__bf16*)(ws + 56 * MB);   // 16MB
    __bf16* vbuf   = (__bf16*)(ws + 72 * MB);   // 16MB (cross V^T uses 8MB)
    __bf16* qbuf   = (__bf16*)(ws + 88 * MB);   // 16MB
    __bf16* attacc = (__bf16*)(ws + 104 * MB);  // 16MB
    float*  x1     = (float*)(ws + 120 * MB);   // 32MB
    __bf16* hbuf   = (__bf16*)(ws + 56 * MB);   // 64MB, overlays dead k/v/q/attacc
    __bf16* sobuf  = (__bf16*)(ws + 152 * MB);  // 8KB
    float*  sproj  = (float*)(ws + 152 * MB + 64 * 1024);  // 16KB

    // 0) weights -> bf16 transposed; encoding -> bf16
    for (int i = 0; i < 10; i++) {
        size_t tot = (size_t)wK[i] * wN[i];
        convT_kernel<<<dim3((unsigned)((tot + 255) / 256)), 256, 0, stream>>>(Wsrc[i], WT[i], wK[i], wN[i]);
    }
    conv_kernel<<<dim3(4096u * 4u), 256, 0, stream>>>(enc, encbf, (size_t)4096 * 1024);

    // 1) self-attention branch
    rmsnorm_kernel<<<dim3(8192), 256, 0, stream>>>(x, n1, nbuf);
    gemm_kernel<1, false><<<dim3(64, 16), 256, 0, stream>>>(nbuf, WT[1], kbuf, nullptr, 8192, 1024, 1024, 0);
    gemm_kernel<1, false><<<dim3(64, 16), 256, 0, stream>>>(nbuf, WT[2], vbuf, nullptr, 8192, 1024, 1024, 0);
    self_attn_kernel<<<dim3(64), 256, 0, stream>>>(nbuf, WT[0], kbuf, vbuf, bias_s, sobuf);
    gemm_kernel<0, false><<<dim3(1, 16), 256, 0, stream>>>(sobuf, WT[3], sproj, nullptr, 4, 1024, 1024, 0);
    bcast_add_kernel<<<dim3(32768), 256, 0, stream>>>(x, sproj, x1);

    // 2) cross-attention branch
    rmsnorm_kernel<<<dim3(8192), 256, 0, stream>>>(x1, n2, nbuf);
    gemm_kernel<1, false><<<dim3(64, 16), 256, 0, stream>>>(nbuf, WT[4], qbuf, nullptr, 8192, 1024, 1024, 0);
    gemm_kernel<1, false><<<dim3(32, 16), 256, 0, stream>>>(encbf, WT[5], kbuf, nullptr, 4096, 1024, 1024, 0);
    gemm_kernel<1, true><<<dim3(32, 16), 256, 0, stream>>>(encbf, WT[6], vbuf, nullptr, 4096, 1024, 1024, 1024);
    flash_cross_kernel<<<dim3(1024), 256, 0, stream>>>(qbuf, kbuf, vbuf, bias_c, attacc);
    gemm_kernel<3, false><<<dim3(64, 16), 256, 0, stream>>>(attacc, WT[7], out, x1, 8192, 1024, 1024, 0);

    // 3) MLP branch (out currently holds x2; final result = x2 + mlp)
    rmsnorm_kernel<<<dim3(8192), 256, 0, stream>>>(out, n3, nbuf);
    gemm_kernel<2, false><<<dim3(64, 64), 256, 0, stream>>>(nbuf, WT[8], hbuf, nullptr, 8192, 4096, 1024, 0);
    gemm_kernel<3, false><<<dim3(64, 16), 256, 0, stream>>>(hbuf, WT[9], out, out, 8192, 1024, 4096, 0);
}
